// FakeT5Attention_17111149707447
// MI455X (gfx1250) — matrix operand order repaired and checked by reading, not compiled, not run on hardware
//
#include <hip/hip_runtime.h>

#define B_ 4
#define S_ 2048
#define D_ 1024
#define H_ 16
#define DK_ 64
#define R_ 16
#define SCALING_ 2.0f

typedef __bf16 bf16;
typedef bf16  v16bf __attribute__((ext_vector_type(16)));
typedef bf16  v8bf  __attribute__((ext_vector_type(8)));
typedef float v8f   __attribute__((ext_vector_type(8)));

__device__ __forceinline__ bf16 to_bf16(float f) {
  union { float f; unsigned u; } c; c.f = f;
  unsigned r = c.u + 0x7FFFu + ((c.u >> 16) & 1u);
  unsigned short hh = (unsigned short)(r >> 16);
  union { unsigned short s; bf16 b; } o; o.s = hh;
  return o.b;
}

// A fragment 16(M)x32(K) bf16: lanes0-15 row M=lane, K {0..7,16..23};
// lanes16-31 same rows' K {8..15,24..31}  (ISA 7.12.2, 16-bit A 16x32)
__device__ __forceinline__ v16bf load_fragA(const bf16* base, int ld, int row0, int k0) {
  const int lane = threadIdx.x & 31;
  const bf16* p = base + (size_t)(row0 + (lane & 15)) * ld + k0 + ((lane >> 4) << 3);
  v8bf lo = *(const v8bf*)p;
  v8bf hi = *(const v8bf*)(p + 16);
  v16bf f;
#pragma unroll
  for (int i = 0; i < 8; ++i) { f[i] = lo[i]; f[i + 8] = hi[i]; }
  return f;
}

// B fragment 32(K)x16(N) bf16: column n = lane%16; lanes0-15 K {0..7,16..23}, lanes16-31 K {8..15,24..31}.
// Source stored row-per-column (W rows / K rows / V dk-rows), contiguous over K.
__device__ __forceinline__ v16bf load_fragB(const bf16* base, int ld, int n0, int k0) {
  const int lane = threadIdx.x & 31;
  const bf16* p = base + (size_t)(n0 + (lane & 15)) * ld + k0 + ((lane >> 4) << 3);
  v8bf lo = *(const v8bf*)p;
  v8bf hi = *(const v8bf*)(p + 16);
  v16bf f;
#pragma unroll
  for (int i = 0; i < 8; ++i) { f[i] = lo[i]; f[i + 8] = hi[i]; }
  return f;
}

#define WMMA_BF16(a, b, c) \
  __builtin_amdgcn_wmma_f32_16x16x32_bf16(false, (a), false, (b), (short)0, (c), false, false)

__global__ void convert_kernel(const float* __restrict__ in, bf16* __restrict__ out, int n) {
  int i = blockIdx.x * blockDim.x + threadIdx.x;
  if (i < n) out[i] = to_bf16(in[i]);
}

// Weff[i][j] = W[i][j] + SCALING * sum_r Bm[i][r] * A[r][j], stored bf16
__global__ void fold_lora_kernel(const float* __restrict__ W, const float* __restrict__ A,
                                 const float* __restrict__ Bm, bf16* __restrict__ out) {
  int idx = blockIdx.x * blockDim.x + threadIdx.x;
  if (idx >= D_ * D_) return;
  int i = idx >> 10, j = idx & 1023;
  float s = 0.0f;
#pragma unroll
  for (int r = 0; r < R_; ++r) s += Bm[i * R_ + r] * A[r * D_ + j];
  out[idx] = to_bf16(W[idx] + SCALING_ * s);
}

// Y = X @ W^T.  Wave tile 32(M) x 64(N), block tile 64 x 128.
// mode 0: f32 row-major out. mode 1: bf16 head-layout [BH,S,64]. mode 2: bf16 dk-major [BH,64,S].
__global__ void gemm_bf16_kernel(const bf16* __restrict__ X, const bf16* __restrict__ W,
                                 void* __restrict__ outp, int mode, int N, int K) {
  const int lane = threadIdx.x & 31;
  const int wv   = threadIdx.x >> 5;
  const int half = lane >> 4;
  const int cn   = lane & 15;
  const int nblk = blockIdx.x * 128 + (wv & 1) * 64;
  const int mblk = blockIdx.y * 64  + (wv >> 1) * 32;

  v8f acc[2][4];
#pragma unroll
  for (int i = 0; i < 2; ++i)
#pragma unroll
    for (int j = 0; j < 4; ++j) acc[i][j] = {};

  for (int k0 = 0; k0 < K; k0 += 32) {
    v16bf a0 = load_fragA(X, K, mblk,      k0);
    v16bf a1 = load_fragA(X, K, mblk + 16, k0);
    v16bf bfr[4];
#pragma unroll
    for (int j = 0; j < 4; ++j) bfr[j] = load_fragB(W, K, nblk + j * 16, k0);
#pragma unroll
    for (int j = 0; j < 4; ++j) {
      acc[0][j] = WMMA_BF16(a0, bfr[j], acc[0][j]);
      acc[1][j] = WMMA_BF16(a1, bfr[j], acc[1][j]);
    }
  }

#pragma unroll
  for (int i = 0; i < 2; ++i) {
#pragma unroll
    for (int j = 0; j < 4; ++j) {
#pragma unroll
      for (int e = 0; e < 8; ++e) {
        int row = mblk + i * 16 + e + half * 8;
        int col = nblk + j * 16 + cn;
        float val = acc[i][j][e];
        if (mode == 0) {
          ((float*)outp)[(size_t)row * N + col] = val;
        } else {
          int b = row >> 11, s = row & (S_ - 1);
          int h = col >> 6,  dk = col & (DK_ - 1);
          size_t off = (mode == 1)
            ? ((size_t)(b * H_ + h) * S_ + s) * DK_ + dk
            : ((size_t)(b * H_ + h) * DK_ + dk) * S_ + s;
          ((bf16*)outp)[off] = to_bf16(val);
        }
      }
    }
  }
}

// Flash attention: one wave = 16 queries of one (b,h); streams 32 keys/iter.
// LDS probability bounce is per-wave; LDS ops are in-order per wave (ISA 7.3),
// so only a compiler reordering fence (wave_barrier) is needed — no s_barrier.
__global__ void attn_kernel(const bf16* __restrict__ qb, const bf16* __restrict__ kb,
                            const bf16* __restrict__ vb, const float* __restrict__ bias,
                            bf16* __restrict__ ob) {
  __shared__ __align__(16) bf16 pls[4][16][32];
  const int lane = threadIdx.x & 31;
  const int wv   = threadIdx.x >> 5;
  const int half = lane >> 4;
  const int cn   = lane & 15;
  const int bh = blockIdx.y;
  const int b = bh >> 4, h = bh & 15;
  const int m0 = (blockIdx.x * 4 + wv) * 16;

  const bf16* qp = qb + (size_t)bh * S_ * DK_;
  const bf16* kp = kb + (size_t)bh * S_ * DK_;
  const bf16* vp = vb + (size_t)bh * DK_ * S_;
  const float* bp = bias + (size_t)h * S_ * S_;

  v16bf aq0 = load_fragA(qp, DK_, m0, 0);
  v16bf aq1 = load_fragA(qp, DK_, m0, 32);

  v8f o0 = {}, o1 = {}, o2 = {}, o3 = {};
  float mx[8], sm[8];
#pragma unroll
  for (int e = 0; e < 8; ++e) { mx[e] = -3.0e38f; sm[e] = 0.0f; }

  for (int kt = 0; kt < S_; kt += 32) {
    v16bf bk0a = load_fragB(kp, DK_, kt,      0);
    v16bf bk0b = load_fragB(kp, DK_, kt,      32);
    v16bf bk1a = load_fragB(kp, DK_, kt + 16, 0);
    v16bf bk1b = load_fragB(kp, DK_, kt + 16, 32);
    v8f s0 = {}, s1 = {};
    s0 = WMMA_BF16(aq0, bk0a, s0);
    s0 = WMMA_BF16(aq1, bk0b, s0);
    s1 = WMMA_BF16(aq0, bk1a, s1);
    s1 = WMMA_BF16(aq1, bk1b, s1);

#pragma unroll
    for (int e = 0; e < 8; ++e) {
      int qrow = m0 + e + half * 8;
      const float* bq = bp + (size_t)qrow * S_ + kt + cn;
      s0[e] += bq[0];
      s1[e] += bq[16];
      __builtin_prefetch(bq + 32, 0, 1);   // next 32-key tile of bias
    }

#pragma unroll
    for (int e = 0; e < 8; ++e) {
      float rv = fmaxf(s0[e], s1[e]);
#pragma unroll
      for (int msk = 1; msk < 16; msk <<= 1) rv = fmaxf(rv, __shfl_xor(rv, msk, 32));
      float nm   = fmaxf(mx[e], rv);
      float corr = __expf(mx[e] - nm);
      mx[e] = nm;
      float p0 = __expf(s0[e] - nm);
      float p1 = __expf(s1[e] - nm);
      float ps = p0 + p1;
#pragma unroll
      for (int msk = 1; msk < 16; msk <<= 1) ps += __shfl_xor(ps, msk, 32);
      sm[e] = sm[e] * corr + ps;
      o0[e] *= corr; o1[e] *= corr; o2[e] *= corr; o3[e] *= corr;
      int rr = e + half * 8;
      pls[wv][rr][cn]      = to_bf16(p0);
      pls[wv][rr][16 + cn] = to_bf16(p1);
    }
    __builtin_amdgcn_wave_barrier();   // keep LDS stores before the reads below

    v16bf ap;
    {
      const bf16* pp = &pls[wv][cn][half * 8];
      v8bf lo = *(const v8bf*)pp;
      v8bf hi = *(const v8bf*)(pp + 16);
#pragma unroll
      for (int i = 0; i < 8; ++i) { ap[i] = lo[i]; ap[i + 8] = hi[i]; }
    }
    o0 = WMMA_BF16(ap, load_fragB(vp, S_, 0,  kt), o0);
    o1 = WMMA_BF16(ap, load_fragB(vp, S_, 16, kt), o1);
    o2 = WMMA_BF16(ap, load_fragB(vp, S_, 32, kt), o2);
    o3 = WMMA_BF16(ap, load_fragB(vp, S_, 48, kt), o3);
    __builtin_amdgcn_wave_barrier();   // keep reads before next iteration's stores
  }

#pragma unroll
  for (int e = 0; e < 8; ++e) {
    float inv = 1.0f / sm[e];
    int s = m0 + e + half * 8;
    size_t rowoff = ((size_t)b * S_ + s) * (H_ * DK_) + (size_t)h * DK_ + cn;
    ob[rowoff +  0] = to_bf16(o0[e] * inv);
    ob[rowoff + 16] = to_bf16(o1[e] * inv);
    ob[rowoff + 32] = to_bf16(o2[e] * inv);
    ob[rowoff + 48] = to_bf16(o3[e] * inv);
  }
}

extern "C" void kernel_launch(void* const* d_in, const int* in_sizes, int n_in,
                              void* d_out, int out_size, void* d_ws, size_t ws_size,
                              hipStream_t stream) {
  (void)in_sizes; (void)n_in; (void)out_size; (void)ws_size;
  const float* x    = (const float*)d_in[0];
  const float* bias = (const float*)d_in[1];
  const float* Wq   = (const float*)d_in[2];
  const float* Wk   = (const float*)d_in[3];
  const float* Wv   = (const float*)d_in[4];
  const float* Wo   = (const float*)d_in[5];
  const float* Aq   = (const float*)d_in[6];
  const float* Bq   = (const float*)d_in[7];
  const float* Av   = (const float*)d_in[8];
  const float* Bv   = (const float*)d_in[9];
  float* out = (float*)d_out;

  const size_t XE = (size_t)B_ * S_ * D_;   // 8,388,608
  const size_t WE = (size_t)D_ * D_;        // 1,048,576
  bf16* xb  = (bf16*)d_ws;
  bf16* wqe = xb  + XE;
  bf16* wke = wqe + WE;
  bf16* wve = wke + WE;
  bf16* woe = wve + WE;
  bf16* qb  = woe + WE;
  bf16* kb  = qb  + XE;
  bf16* vb  = kb  + XE;
  bf16* ob  = vb  + XE;

  convert_kernel<<<(int)((XE + 255) / 256), 256, 0, stream>>>(x,  xb,  (int)XE);
  convert_kernel<<<(int)((WE + 255) / 256), 256, 0, stream>>>(Wk, wke, (int)WE);
  convert_kernel<<<(int)((WE + 255) / 256), 256, 0, stream>>>(Wo, woe, (int)WE);
  fold_lora_kernel<<<(int)((WE + 255) / 256), 256, 0, stream>>>(Wq, Aq, Bq, wqe);
  fold_lora_kernel<<<(int)((WE + 255) / 256), 256, 0, stream>>>(Wv, Av, Bv, wve);

  dim3 ggrid(D_ / 128, (B_ * S_) / 64);
  gemm_bf16_kernel<<<ggrid, 128, 0, stream>>>(xb, wqe, (void*)qb, 1, D_, D_);
  gemm_bf16_kernel<<<ggrid, 128, 0, stream>>>(xb, wke, (void*)kb, 1, D_, D_);
  gemm_bf16_kernel<<<ggrid, 128, 0, stream>>>(xb, wve, (void*)vb, 2, D_, D_);

  attn_kernel<<<dim3(S_ / 64, B_ * H_), 128, 0, stream>>>(qb, kb, vb, bias, ob);

  gemm_bf16_kernel<<<ggrid, 128, 0, stream>>>(ob, woe, (void*)out, 0, D_, D_);
}
